// InterpretableMultiHeadAttention_78950088835183
// MI455X (gfx1250) — matrix operand order repaired and checked by reading, not compiled, not run on hardware
//
#include <hip/hip_runtime.h>
#include <hip/hip_bf16.h>
#include <cstdint>

// ---------------- problem constants ----------------
#define BB    4
#define SQL   2048
#define SKL   2048
#define DD    1024
#define HH    16
#define HDD   64
#define MTOT  (BB * SQL)        // 8192 token rows
#define SCALE_F 0.125f          // 1/sqrt(64)
#define SKP   2056              // padded LDS row stride (floats)

typedef __bf16 bf16_t;
typedef __bf16 v16bf __attribute__((ext_vector_type(16)));
typedef float  v8f   __attribute__((ext_vector_type(8)));
typedef int    v4i   __attribute__((vector_size(16)));   // matches builtin param

// ---------------- CDNA5 async copy to LDS (guarded) ----------------
#if __has_builtin(__builtin_amdgcn_global_load_async_to_lds_b128)
#define HAVE_ASYNC_LDS 1
#endif

__device__ __forceinline__ void async_copy16(const bf16_t* g, bf16_t* l) {
#if defined(HAVE_ASYNC_LDS)
  __builtin_amdgcn_global_load_async_to_lds_b128(
      (__attribute__((address_space(1))) v4i*)g,
      (__attribute__((address_space(3))) v4i*)l, 0, 0);
#else
  *(uint4*)l = *(const uint4*)g;
#endif
}

__device__ __forceinline__ void async_wait0() {
#if defined(HAVE_ASYNC_LDS)
#if __has_builtin(__builtin_amdgcn_s_wait_asynccnt)
  __builtin_amdgcn_s_wait_asynccnt(0);
#else
  asm volatile("s_wait_asynccnt 0" ::: "memory");
#endif
#endif
}

// ---------------- WMMA helpers ----------------
__device__ __forceinline__ v8f wmma_bf16(v16bf a, v16bf b, v8f c) {
  return __builtin_amdgcn_wmma_f32_16x16x32_bf16(false, a, false, b,
                                                 (short)0, c, false, false);
}

// A fragment: lane holds K = {0..7} and {16..23} (lanes 0-15) / +8 (lanes 16-31)
__device__ __forceinline__ v16bf load_frag_a(const bf16_t* p0, const bf16_t* p1) {
  union { v16bf v; uint4 u[2]; } f;
  f.u[0] = *(const uint4*)p0;
  f.u[1] = *(const uint4*)p1;
  return f.v;
}
// B fragment: lane holds K = {0..7} and {16..23} (lanes 0-15) / +8 (lanes 16-31)
__device__ __forceinline__ v16bf load_frag_b32(const bf16_t* p) {
  union { v16bf v; uint4 u[2]; } f;
  f.u[0] = *(const uint4*)p;
  f.u[1] = *(const uint4*)(p + 16);
  return f.v;
}

// ---------------- elementwise convert / weight-reduce ----------------
__global__ void cvt_f32_bf16_kernel(const float* __restrict__ x,
                                    bf16_t* __restrict__ y, int n4) {
  int i = blockIdx.x * blockDim.x + threadIdx.x;
  if (i >= n4) return;
  float4 v = ((const float4*)x)[i];
  union { ushort4 u; bf16_t b[4]; } p;
  p.b[0] = (bf16_t)v.x; p.b[1] = (bf16_t)v.y;
  p.b[2] = (bf16_t)v.z; p.b[3] = (bf16_t)v.w;
  ((ushort4*)y)[i] = p.u;
}

// Wo_red[n][d] = sum_h Wo[n][h*64+d]  (tile(x,H) @ Wo^T == x @ Wo_red^T)
__global__ void wo_reduce_kernel(const float* __restrict__ Wo,
                                 bf16_t* __restrict__ Wr) {
  int i = blockIdx.x * blockDim.x + threadIdx.x;   // 0 .. 1024*64-1
  int n = i >> 6, d = i & 63;
  float s = 0.f;
#pragma unroll
  for (int h = 0; h < HH; ++h) s += Wo[n * DD + h * HDD + d];
  Wr[i] = (bf16_t)s;
}

// ---------------- projection GEMM: Y = X @ W^T + b  (bf16 WMMA) ----------------
// X:(M,1024) bf16, W:(1024,1024) bf16 row-major (N,K), Out scattered per head.
// mode 0: Out[((b*H+h)*S + s)*64 + d]   (Q,K layout)
// mode 1: Out[((b*H+h)*64 + d)*S + s]   (V transposed for p@V B-fragments)
__global__ __launch_bounds__(256) void proj_gemm(
    const bf16_t* __restrict__ X, const bf16_t* __restrict__ Wt,
    const float* __restrict__ bias, bf16_t* __restrict__ Out, int mode) {
  __shared__ bf16_t As[128 * 32];
  __shared__ bf16_t Bs[128 * 32];
  const int tid = threadIdx.x;
  const int w = tid >> 5, l = tid & 31;
  const int lrow = l & 15;
  const int ha = (l >> 4) * 8;      // A-frag K offset
  const int hb = (l >> 4) * 8;      // B-frag K offset
  const int rb = (l >> 4) << 3;     // C/D row base
  const int M0 = blockIdx.x * 128;
  const int N0 = blockIdx.y * 128;
  const int wm = (w & 1) * 64;
  const int wn = (w >> 1) * 32;

  v8f acc[4][2] = {};

  for (int kk = 0; kk < DD; kk += 32) {
#pragma unroll
    for (int i = 0; i < 2; ++i) {                 // 512 16B chunks / 256 thr
      int cch = tid + i * 256;
      int row = cch >> 2, seg = cch & 3;
      async_copy16(X  + (size_t)(M0 + row) * DD + kk + seg * 8,
                   &As[row * 32 + seg * 8]);
      async_copy16(Wt + (size_t)(N0 + row) * DD + kk + seg * 8,
                   &Bs[row * 32 + seg * 8]);
    }
    async_wait0();
    __syncthreads();

    v16bf af[4], bfr[2];
#pragma unroll
    for (int i = 0; i < 4; ++i) {
      const bf16_t* p = &As[(wm + 16 * i + lrow) * 32];
      af[i] = load_frag_a(p + ha, p + ha + 16);
    }
#pragma unroll
    for (int j = 0; j < 2; ++j)
      bfr[j] = load_frag_b32(&Bs[(wn + 16 * j + lrow) * 32 + hb]);
#pragma unroll
    for (int i = 0; i < 4; ++i)
#pragma unroll
      for (int j = 0; j < 2; ++j)
        acc[i][j] = wmma_bf16(af[i], bfr[j], acc[i][j]);
    __syncthreads();
  }

#pragma unroll
  for (int i = 0; i < 4; ++i)
#pragma unroll
    for (int j = 0; j < 2; ++j) {
      int n = N0 + wn + 16 * j + lrow;
      float bvv = bias[n];
      int hh = n >> 6, d = n & 63;
#pragma unroll
      for (int r = 0; r < 8; ++r) {
        int m = M0 + wm + 16 * i + rb + r;
        int bb = m >> 11, s = m & 2047;
        size_t idx = (mode == 0)
            ? ((size_t)(bb * HH + hh) * SQL + s) * HDD + d
            : ((size_t)(bb * HH + hh) * HDD + d) * (size_t)SKL + s;
        Out[idx] = (bf16_t)(acc[i][j][r] + bvv);
      }
    }
}

// ---------------- fused attention over all heads ----------------
// WG = (b, 16 q-rows); LDS holds the 16x2048 score block and the 16x2048
// head-averaged attention accumulator; 16x64 attended accumulator stays in
// VGPRs across the head loop (mean over heads applied at the end).
__global__ __launch_bounds__(256) void attn_kernel(
    const bf16_t* __restrict__ Qb,   // (B,H,SQ,64)
    const bf16_t* __restrict__ Kb,   // (B,H,SK,64)
    const bf16_t* __restrict__ Vtb,  // (B,H,64,SK)
    float* __restrict__ attOut,      // (B,SQ,64)  head-mean attended
    float* __restrict__ avgOut) {    // (B,SQ,SK)  head-mean attention
  extern __shared__ char smem[];
  float*  sc  = (float*)smem;              // [16][SKP]
  float*  avg = sc + 16 * SKP;             // [16][SKP]
  bf16_t* qt  = (bf16_t*)(avg + 16 * SKP); // [16][64]

  const int tid = threadIdx.x;
  const int w = tid >> 5, l = tid & 31;
  const int lrow = l & 15;
  const int ha = (l >> 4) * 8;
  const int hb = (l >> 4) * 8;
  const int rb = (l >> 4) << 3;
  const int q0 = blockIdx.x * 16;
  const int b  = blockIdx.y;
  const float invH = 1.0f / (float)HH;

  for (int i = tid; i < 16 * SKP; i += 256) avg[i] = 0.0f;

  v8f oacc = {};
  for (int h = 0; h < HH; ++h) {
    if (tid < 128) {                                   // stage Q tile (16x64)
      int row = tid >> 3, seg = tid & 7;
      const bf16_t* g =
          Qb + ((size_t)(b * HH + h) * SQL + q0 + row) * HDD + seg * 8;
      *(uint4*)&qt[row * HDD + seg * 8] = *(const uint4*)g;
    }
    __syncthreads();

    // ---- scores: wave w covers key columns [w*256, w*256+256) ----
    const bf16_t* qr = &qt[lrow * HDD];
    v16bf a0 = load_frag_a(qr + ha, qr + ha + 16);
    v16bf a1 = load_frag_a(qr + 32 + ha, qr + 32 + ha + 16);
    const bf16_t* Kh = Kb + (size_t)(b * HH + h) * SKL * HDD;
#pragma unroll 4
    for (int t = 0; t < 16; ++t) {
      int kcol = w * 256 + t * 16;
      const bf16_t* kr = Kh + (size_t)(kcol + lrow) * HDD;
      v16bf b0 = load_frag_b32(kr + hb);
      v16bf b1 = load_frag_b32(kr + 32 + hb);
      v8f c = {};
      c = wmma_bf16(a0, b0, c);
      c = wmma_bf16(a1, b1, c);
#pragma unroll
      for (int r = 0; r < 8; ++r)
        sc[(rb + r) * SKP + kcol + lrow] = c[r] * SCALE_F;
    }
    __syncthreads();

    // ---- exact softmax: wave w owns rows 2w, 2w+1 ----
#pragma unroll
    for (int rr = 0; rr < 2; ++rr) {
      int row = 2 * w + rr;
      float* srow = &sc[row * SKP];
      float* arow = &avg[row * SKP];
      float m = -3.0e38f;
      for (int c = l; c < SKL; c += 32) m = fmaxf(m, srow[c]);
#pragma unroll
      for (int off = 16; off > 0; off >>= 1) m = fmaxf(m, __shfl_xor(m, off, 32));
      float s = 0.0f;
      for (int c = l; c < SKL; c += 32) {
        float e = __expf(srow[c] - m);
        srow[c] = e; s += e;
      }
#pragma unroll
      for (int off = 16; off > 0; off >>= 1) s += __shfl_xor(s, off, 32);
      float inv = 1.0f / s;
      for (int c = l; c < SKL; c += 32) {
        float p = srow[c] * inv;
        srow[c] = p;
        arow[c] += p * invH;        // head-average accumulator
      }
    }
    __syncthreads();

    // ---- p @ V : waves 0..3 each own 16 of the 64 output dims ----
    if (w < 4) {
      const bf16_t* vrow =
          Vtb + ((size_t)(b * HH + h) * HDD + (w * 16 + lrow)) * (size_t)SKL;
      const float* prow = &sc[lrow * SKP];
      for (int kt = 0; kt < SKL; kt += 32) {
        v16bf a;
        const float* pr = prow + kt + ha;
#pragma unroll
        for (int e = 0; e < 8; ++e) a[e] = (bf16_t)pr[e];
#pragma unroll
        for (int e = 0; e < 8; ++e) a[8 + e] = (bf16_t)pr[16 + e];
        v16bf bv = load_frag_b32(vrow + kt + hb);
        oacc = wmma_bf16(a, bv, oacc);     // accumulates across heads too
      }
    }
    __syncthreads();
  }

  if (w < 4) {
#pragma unroll
    for (int r = 0; r < 8; ++r)
      attOut[((size_t)b * SQL + q0 + rb + r) * HDD + w * 16 + lrow] =
          oacc[r] * invH;
  }

  float* aOut = avgOut + ((size_t)b * SQL + q0) * (size_t)SKL;
  for (int i = tid; i < 16 * (SKL / 4); i += 256) {
    int row = i / (SKL / 4), c4 = i % (SKL / 4);
    *(float4*)&aOut[(size_t)row * SKL + c4 * 4] =
        *(float4*)&avg[row * SKP + c4 * 4];
  }
}

// ---------------- output projection: out = att @ Wo_red^T + bo (K=64) --------
__global__ __launch_bounds__(256) void out_proj(
    const float* __restrict__ att,   // (8192,64) f32
    const bf16_t* __restrict__ Wr,   // (1024,64) bf16
    const float* __restrict__ bo, float* __restrict__ out) {
  const int tid = threadIdx.x;
  const int w = tid >> 5, l = tid & 31;
  const int lrow = l & 15;
  const int ha = (l >> 4) * 8;
  const int hb = (l >> 4) * 8;
  const int rb = (l >> 4) << 3;
  const int M0 = blockIdx.x * 128;
  const int N0 = blockIdx.y * 128;
  const int wm = (w & 1) * 64;
  const int wn = (w >> 1) * 32;

  v8f acc[4][2] = {};
#pragma unroll
  for (int kk = 0; kk < HDD; kk += 32) {
    v16bf af[4], bfr[2];
#pragma unroll
    for (int i = 0; i < 4; ++i) {
      const float* pr = att + (size_t)(M0 + wm + 16 * i + lrow) * HDD + kk + ha;
      v16bf a;
#pragma unroll
      for (int e = 0; e < 8; ++e) a[e] = (bf16_t)pr[e];
#pragma unroll
      for (int e = 0; e < 8; ++e) a[8 + e] = (bf16_t)pr[16 + e];
      af[i] = a;
    }
#pragma unroll
    for (int j = 0; j < 2; ++j)
      bfr[j] = load_frag_b32(Wr + (size_t)(N0 + wn + 16 * j + lrow) * HDD + kk + hb);
#pragma unroll
    for (int i = 0; i < 4; ++i)
#pragma unroll
      for (int j = 0; j < 2; ++j)
        acc[i][j] = wmma_bf16(af[i], bfr[j], acc[i][j]);
  }
#pragma unroll
  for (int i = 0; i < 4; ++i)
#pragma unroll
    for (int j = 0; j < 2; ++j) {
      int n = N0 + wn + 16 * j + lrow;
      float bv = bo[n];
#pragma unroll
      for (int r = 0; r < 8; ++r) {
        int m = M0 + wm + 16 * i + rb + r;
        out[(size_t)m * DD + n] = acc[i][j][r] + bv;
      }
    }
}

// ---------------- launcher ----------------
extern "C" void kernel_launch(void* const* d_in, const int* in_sizes, int n_in,
                              void* d_out, int out_size, void* d_ws,
                              size_t ws_size, hipStream_t stream) {
  const float* query = (const float*)d_in[0];
  const float* key_  = (const float*)d_in[1];
  const float* value = (const float*)d_in[2];
  const float* Wq = (const float*)d_in[3];
  const float* bq = (const float*)d_in[4];
  const float* Wk = (const float*)d_in[5];
  const float* bk = (const float*)d_in[6];
  const float* Wv = (const float*)d_in[7];
  const float* bv = (const float*)d_in[8];
  const float* Wo = (const float*)d_in[9];
  const float* bo = (const float*)d_in[10];
  float* out = (float*)d_out;

  char* ws = (char*)d_ws;
  size_t off = 0;
  auto carve = [&](size_t bytes) -> char* {
    off = (off + 255) & ~(size_t)255;
    char* p = ws + off;
    off += bytes;
    return p;
  };
  const size_t act_e = (size_t)MTOT * DD;           // 8.39M elems
  bf16_t* Qb  = (bf16_t*)carve(act_e * 2);          // (B,H,SQ,64)
  bf16_t* Kb  = (bf16_t*)carve(act_e * 2);          // (B,H,SK,64)
  bf16_t* Vtb = (bf16_t*)carve(act_e * 2);          // (B,H,64,SK)
  float*  attw = (float*)carve((size_t)MTOT * HDD * 4);  // (B,SQ,64)
  bf16_t* Xq = (bf16_t*)carve(act_e * 2);
  bf16_t* Xk = (bf16_t*)carve(act_e * 2);
  bf16_t* Xv = (bf16_t*)carve(act_e * 2);
  bf16_t* Wqb = (bf16_t*)carve((size_t)DD * DD * 2);
  bf16_t* Wkb = (bf16_t*)carve((size_t)DD * DD * 2);
  bf16_t* Wvb = (bf16_t*)carve((size_t)DD * DD * 2);
  bf16_t* Wob = (bf16_t*)carve((size_t)DD * HDD * 2);

  auto cvt = [&](const float* s, bf16_t* d, size_t n) {
    int n4 = (int)(n / 4);
    cvt_f32_bf16_kernel<<<(n4 + 255) / 256, 256, 0, stream>>>(s, d, n4);
  };
  cvt(query, Xq, act_e);
  cvt(key_,  Xk, act_e);
  cvt(value, Xv, act_e);
  cvt(Wq, Wqb, (size_t)DD * DD);
  cvt(Wk, Wkb, (size_t)DD * DD);
  cvt(Wv, Wvb, (size_t)DD * DD);
  wo_reduce_kernel<<<(DD * HDD) / 256, 256, 0, stream>>>(Wo, Wob);

  dim3 pgrid(MTOT / 128, DD / 128);                 // 64 x 8
  proj_gemm<<<pgrid, 256, 0, stream>>>(Xq, Wqb, bq, Qb, 0);
  proj_gemm<<<pgrid, 256, 0, stream>>>(Xk, Wkb, bk, Kb, 0);
  proj_gemm<<<pgrid, 256, 0, stream>>>(Xv, Wvb, bv, Vtb, 1);

  size_t attn_lds = (size_t)2 * 16 * SKP * sizeof(float) +
                    (size_t)16 * HDD * sizeof(bf16_t);   // ~259 KB < 320 KB
  attn_kernel<<<dim3(SQL / 16, BB), 256, attn_lds, stream>>>(
      Qb, Kb, Vtb, attw, out + (size_t)MTOT * DD);

  out_proj<<<dim3(MTOT / 128, DD / 128), 256, 0, stream>>>(attw, Wob, bo, out);

  (void)in_sizes; (void)n_in; (void)out_size; (void)ws_size;
}
